// NaryCompUSchema_32633161515450
// MI455X (gfx1250) — matrix operand order repaired and checked by reading, not compiled, not run on hardware
//
#include <hip/hip_runtime.h>
#include <hip/hip_bf16.h>

// ---------------------------------------------------------------------------
// Problem constants (from the reference)
// ---------------------------------------------------------------------------
#define NSENT   4096
#define EMB     300
#define EMB_PAD 320              // EMB padded to a multiple of 32 (WMMA K step)
#define HID     256
#define GATES   1024             // 4*HID
#define KTOT    (EMB_PAD + HID)  // 576: fused K dim = [embedding | hidden]
#define NIT     (KTOT / 32)      // 18 K-iterations
#define IT_TAIL 9                // iteration with the k<300 guard (K 288..319)
#define T_COL   64
#define T_ROW   16

typedef __attribute__((ext_vector_type(16))) __bf16 v16bf;
typedef __attribute__((ext_vector_type(8)))  __bf16 v8bf;
typedef __attribute__((ext_vector_type(8)))  float  v8f;

__device__ __forceinline__ v8bf cvt8(float4 a, float4 b) {
    v8bf r;
    r[0] = (__bf16)a.x; r[1] = (__bf16)a.y; r[2] = (__bf16)a.z; r[3] = (__bf16)a.w;
    r[4] = (__bf16)b.x; r[5] = (__bf16)b.y; r[6] = (__bf16)b.z; r[7] = (__bf16)b.w;
    return r;
}

// ---------------------------------------------------------------------------
// Prepack weights to bf16 in the fused-K layout the WMMA B fragment reads:
// Wp[j][k] for k in [0,576): [ W_ih[j][0..299] | 0-pad | W_hh[j][0..255] ].
// Row-major, k contiguous -> B fragments are plain aligned 32B global loads.
// ---------------------------------------------------------------------------
__global__ __launch_bounds__(256)
void pack_weights(const float* __restrict__ W_ih, const float* __restrict__ W_hh,
                  __bf16* __restrict__ Wp)
{
    const int j = blockIdx.x;                // 0..1023
    for (int k = threadIdx.x; k < KTOT; k += 256) {
        float v = 0.0f;
        if (k < EMB_PAD) {
            if (k < EMB) v = W_ih[j * EMB + k];
        } else {
            v = W_hh[j * HID + (k - EMB_PAD)];
        }
        Wp[(size_t)j * KTOT + k] = (__bf16)v;
    }
}

// ---------------------------------------------------------------------------
// gates[n, 0:1024] = emb(tokens[n,t]) @ W_ih^T + h[n] @ W_hh^T + (b_ih + b_hh)
//
// Block: 128 threads = 4 waves.  Block tile: 32 rows x 128 gate-cols.
// Wave w: m_off = (w&1)*16, n_base = (w>>1)*64 -> 4 WMMAs reusing one A frag.
// A tile staged in LDS in per-lane fragment order (ping-pong buffered):
//   16-bit A 16x32, contiguous k (kernel's order, B matches): lane = (m&15) + 16*(k>>4), elem = k&15.
// B fragments loaded straight from the prepacked bf16 weights in global/L2.
// ---------------------------------------------------------------------------
__global__ __launch_bounds__(128)
void lstm_gates_gemm(const int* __restrict__ tokens, int t, int T,
                     const float* __restrict__ embed,
                     const float* __restrict__ h,
                     const __bf16* __restrict__ Wp,    // [1024][576] bf16
                     const float* __restrict__ b_ih,
                     const float* __restrict__ b_hh,
                     float* __restrict__ gates)        // [4096, 1024]
{
    // [buf][mtile][lane][elem]
    __shared__ __bf16 Asf[2][2][32][16];

    const int tid  = threadIdx.x;
    const int lane = tid & 31;
    const int wave = tid >> 5;       // 0..3
    const int half = lane >> 4;      // 0|1

    const int m_blk  = blockIdx.x * 32;    // sentence rows
    const int n_blk  = blockIdx.y * 128;   // gate columns
    const int m_off  = (wave & 1) * 16;
    const int n_base = (wave >> 1) * 64;

    // ---- A staging assignment: 128 thr -> 32 rows x 4 chunks of 8 k ----
    const int arow = tid >> 2;             // 0..31
    const int acb  = (tid & 3) * 8;        // 0,8,16,24 (local k chunk)
    const int nrow = m_blk + arow;
    const int tok  = tokens[nrow * T + t];
    const float* __restrict__ embRow = embed + (size_t)tok * EMB;
    const float* __restrict__ hRow   = h + (size_t)nrow * HID;

    // fragment-layout destination for this thread's 8-element chunk
    const int s_mtile = arow >> 4;
    const int s_lane  = (arow & 15) + 16 * (acb >> 4);
    const int s_ebase = acb & 8;

    auto stageA = [&](int it, int buf) {
        v8bf pk;
        if (it < IT_TAIL) {                      // pure embedding, no guards
            const int off = it * 32 + acb;
            const float4* p = (const float4*)(embRow + off);
            pk = cvt8(p[0], p[1]);
        } else if (it == IT_TAIL) {              // K 288..319: guard k<300
            float v[8];
            #pragma unroll
            for (int e = 0; e < 8; ++e) {
                const int kg = 288 + acb + e;
                v[e] = (kg < EMB) ? embRow[kg] : 0.0f;
            }
            pk = cvt8(make_float4(v[0], v[1], v[2], v[3]),
                      make_float4(v[4], v[5], v[6], v[7]));
        } else {                                 // pure hidden, no guards
            const int off = (it - 10) * 32 + acb;
            const float4* p = (const float4*)(hRow + off);
            pk = cvt8(p[0], p[1]);
        }
        *(v8bf*)&Asf[buf][s_mtile][s_lane][s_ebase] = pk;
    };

    v8f acc[4] = {v8f{}, v8f{}, v8f{}, v8f{}};

    const int a_mtile = (wave & 1);
    const int bn      = lane & 15;

    stageA(0, 0);
    __syncthreads();

    for (int it = 0; it < NIT; ++it) {
        const int cur = it & 1;
        if (it + 1 < NIT) stageA(it + 1, cur ^ 1);

        // A fragment: one 32B LDS read (2 x ds_load_b128)
        const v16bf afrag = *(const v16bf*)&Asf[cur][a_mtile][lane][0];

        // 4 B fragments straight from prepacked weights (aligned 32B loads)
        const int k0 = it * 32;
        #pragma unroll
        for (int tt = 0; tt < 4; ++tt) {
            const int col = n_blk + n_base + tt * 16 + bn;
            const v16bf bfrag =
                *(const v16bf*)(Wp + (size_t)col * KTOT + k0 + half * 16);
            acc[tt] = __builtin_amdgcn_wmma_f32_16x16x32_bf16(
                          false, afrag, false, bfrag, (short)0, acc[tt],
                          false, false);
        }
        __syncthreads();
    }

    // ---- store D + bias (C/D layout: VGPR v -> M = v + half*8, N = lane&15) ----
    #pragma unroll
    for (int tt = 0; tt < 4; ++tt) {
        const int col  = n_blk + n_base + tt * 16 + bn;
        const float bs = b_ih[col] + b_hh[col];
        #pragma unroll
        for (int v = 0; v < 8; ++v) {
            const int rowg = m_blk + m_off + v + half * 8;
            gates[(size_t)rowg * GATES + col] = acc[tt][v] + bs;
        }
    }
}

// ---------------------------------------------------------------------------
// Pointwise LSTM cell update with packed-sequence masking (t < len).
// Torch gate order: i, f, g, o.
// ---------------------------------------------------------------------------
__global__ __launch_bounds__(256)
void lstm_update(const float* __restrict__ G, const int* __restrict__ lens,
                 int t, float* __restrict__ h, float* __restrict__ c)
{
    const int n = blockIdx.x;
    const int j = threadIdx.x;
    if (t >= lens[n]) return;   // state frozen past sequence length

    const float* g = G + (size_t)n * GATES;
    float gi = g[j];
    float gf = g[HID + j];
    float gg = g[2 * HID + j];
    float go = g[3 * HID + j];

    gi = 1.0f / (1.0f + __expf(-gi));
    gf = 1.0f / (1.0f + __expf(-gf));
    go = 1.0f / (1.0f + __expf(-go));
    gg = tanhf(gg);

    const int idx = n * HID + j;
    const float cn = gf * c[idx] + gi * gg;
    c[idx] = cn;
    h[idx] = go * tanhf(cn);
}

// ---------------------------------------------------------------------------
// segment_sum: acc[refs[n], :] += h[n, :]   (acc is [NSENT, HID], zeroed)
// ---------------------------------------------------------------------------
__global__ __launch_bounds__(256)
void seg_scatter(const float* __restrict__ h, const int* __restrict__ refs,
                 float* __restrict__ acc)
{
    const int n = blockIdx.x;
    const int j = threadIdx.x;
    atomicAdd(&acc[(size_t)refs[n] * HID + j], h[n * HID + j]);
}

// ---------------------------------------------------------------------------
// BPR loss: out[0] = sum_n -log_sigmoid(dot(col,row) - dot(col,neg))
// ---------------------------------------------------------------------------
__global__ __launch_bounds__(256)
void bpr_loss(const float* __restrict__ colh, const float* __restrict__ rowh,
              const float* __restrict__ negh, float* __restrict__ out)
{
    __shared__ float red[256];
    const int n = blockIdx.x * blockDim.x + threadIdx.x;

    const float* ch = colh + (size_t)n * HID;
    const float* rh = rowh + (size_t)n * HID;
    const float* nh = negh + (size_t)n * HID;

    float pos = 0.0f, neg = 0.0f;
    #pragma unroll 4
    for (int j = 0; j < HID; ++j) {
        const float cv = ch[j];
        pos += cv * rh[j];
        neg += cv * nh[j];
    }
    const float x = pos - neg;
    // stable log_sigmoid(x) = min(x,0) - log1p(exp(-|x|))
    const float ls = fminf(x, 0.0f) - log1pf(__expf(-fabsf(x)));

    red[threadIdx.x] = -ls;
    __syncthreads();
    for (int s = 128; s > 0; s >>= 1) {
        if (threadIdx.x < s) red[threadIdx.x] += red[threadIdx.x + s];
        __syncthreads();
    }
    if (threadIdx.x == 0) atomicAdd(out, red[0]);
}

// ---------------------------------------------------------------------------
// Host-side orchestration
// ---------------------------------------------------------------------------
static void run_stream(const int* tokens, int T, const int* lens, const int* refs,
                       const float* embed, const __bf16* Wp,
                       const float* b_ih, const float* b_hh,
                       float* gates, float* h, float* c, float* acc,
                       hipStream_t stream)
{
    hipMemsetAsync(h,   0, (size_t)NSENT * HID * sizeof(float), stream);
    hipMemsetAsync(c,   0, (size_t)NSENT * HID * sizeof(float), stream);
    hipMemsetAsync(acc, 0, (size_t)NSENT * HID * sizeof(float), stream);

    const dim3 grid(NSENT / 32, GATES / 128);
    for (int t = 0; t < T; ++t) {
        lstm_gates_gemm<<<grid, 128, 0, stream>>>(tokens, t, T, embed, h,
                                                  Wp, b_ih, b_hh, gates);
        lstm_update<<<NSENT, 256, 0, stream>>>(gates, lens, t, h, c);
    }
    seg_scatter<<<NSENT, 256, 0, stream>>>(h, refs, acc);
}

extern "C" void kernel_launch(void* const* d_in, const int* in_sizes, int n_in,
                              void* d_out, int out_size, void* d_ws, size_t ws_size,
                              hipStream_t stream)
{
    const int*   col          = (const int*)  d_in[0];
    const int*   row          = (const int*)  d_in[1];
    const int*   row_neg      = (const int*)  d_in[2];
    const int*   col_lens     = (const int*)  d_in[3];
    const int*   row_lens     = (const int*)  d_in[4];
    const int*   row_neg_lens = (const int*)  d_in[5];
    const int*   col_refs     = (const int*)  d_in[6];
    const int*   row_refs     = (const int*)  d_in[7];
    const int*   row_neg_refs = (const int*)  d_in[8];
    const float* embed        = (const float*)d_in[9];
    const float* col_W_ih     = (const float*)d_in[10];
    const float* col_W_hh     = (const float*)d_in[11];
    const float* col_b_ih     = (const float*)d_in[12];
    const float* col_b_hh     = (const float*)d_in[13];
    const float* row_W_ih     = (const float*)d_in[14];
    const float* row_W_hh     = (const float*)d_in[15];
    const float* row_b_ih     = (const float*)d_in[16];
    const float* row_b_hh     = (const float*)d_in[17];

    // workspace layout (floats): gates | h | c | acc_col | acc_row | acc_neg | Wpack
    float* ws      = (float*)d_ws;
    float* gates   = ws;                                   // 4096*1024
    float* h       = gates   + (size_t)NSENT * GATES;      // 4096*256
    float* c       = h       + (size_t)NSENT * HID;
    float* acc_col = c       + (size_t)NSENT * HID;
    float* acc_row = acc_col + (size_t)NSENT * HID;
    float* acc_neg = acc_row + (size_t)NSENT * HID;
    __bf16* Wpack  = (__bf16*)(acc_neg + (size_t)NSENT * HID); // 1024*576 bf16

    // ---- col stream ----
    pack_weights<<<GATES, 256, 0, stream>>>(col_W_ih, col_W_hh, Wpack);
    run_stream(col, T_COL, col_lens, col_refs, embed, Wpack,
               col_b_ih, col_b_hh, gates, h, c, acc_col, stream);

    // ---- row + row_neg streams (same weights) ----
    pack_weights<<<GATES, 256, 0, stream>>>(row_W_ih, row_W_hh, Wpack);
    run_stream(row, T_ROW, row_lens, row_refs, embed, Wpack,
               row_b_ih, row_b_hh, gates, h, c, acc_row, stream);
    run_stream(row_neg, T_ROW, row_neg_lens, row_neg_refs, embed, Wpack,
               row_b_ih, row_b_hh, gates, h, c, acc_neg, stream);

    hipMemsetAsync(d_out, 0, sizeof(float), stream);
    bpr_loss<<<NSENT / 256, 256, 0, stream>>>(acc_col, acc_row, acc_neg,
                                              (float*)d_out);
}
